// DecoderLSTM_5454608465931
// MI455X (gfx1250) — compile-verified, matrix operand order checked by reading, not run on hardware
//
#include <hip/hip_runtime.h>

// ---------------------------------------------------------------------------
// Types for CDNA5 WMMA (wave32, gfx1250)
// ---------------------------------------------------------------------------
typedef __attribute__((ext_vector_type(16))) __bf16    v16bf;
typedef __attribute__((ext_vector_type(8)))  float     v8f;
typedef __attribute__((ext_vector_type(4)))  unsigned  u32x4;
typedef __attribute__((ext_vector_type(2)))  unsigned  u32x2;

union FragU { u32x4 q[2]; v16bf v; };

__device__ __forceinline__ unsigned short f2bf(float f) {
    unsigned u = __float_as_uint(f);
    unsigned r = (u + 0x7FFFu + ((u >> 16) & 1u)) >> 16;   // round-to-nearest-even
    return (unsigned short)r;
}

// Problem constants
#define BATCH   32
#define NREG    196
#define EMBED   512
#define HIDDEN  512
#define SEQ     64
#define VOCAB   32000
#define GATES4  2048

// LDS row stride (ushorts): 40 -> 80B rows. 16B-aligned (b128 ok) and rows
// start 20 banks apart -> the 16 row-starts hit banks {0,4,..,60}: conflict-free.
#define LSTR    40

// ---------------------------------------------------------------------------
// Optional CDNA5 async global->LDS staging (ASYNCcnt-tracked).
// Builtin signature (from compiler diagnostic): pointers to vector_size(16) int.
// ---------------------------------------------------------------------------
#if defined(__has_builtin)
#if __has_builtin(__builtin_amdgcn_global_load_async_to_lds_b128) && \
    __has_builtin(__builtin_amdgcn_s_wait_asynccnt)
#define USE_ASYNC_LDS 1
#endif
#endif

#ifdef USE_ASYNC_LDS
typedef int v4i_t __attribute__((vector_size(16)));
typedef __attribute__((address_space(1))) v4i_t* gv4i_ptr;   // ptr to global v4i
typedef __attribute__((address_space(3))) v4i_t* lv4i_ptr;   // ptr to LDS v4i
__device__ __forceinline__ void async_copy16(const void* g, void* l) {
    // generic->AS1 is value-preserving; generic LDS pointer's low 32 bits are
    // the LDS offset (ISA 10.2 aperture rules), so trunc to AS3 is valid.
    // AS1/AS3 -> generic parameter conversion is implicit on AMDGPU if needed.
    __builtin_amdgcn_global_load_async_to_lds_b128(
        (gv4i_ptr)(unsigned long long)g,
        (lv4i_ptr)(unsigned)(unsigned long long)l,
        0, 0);
}
#endif

// ---------------------------------------------------------------------------
// Generic small WMMA GEMM: C[M x N] = concat_K( A0 @ B0t, A1 @ B1t ) + bias
// A*, B*t bf16 (row-major; B pre-transposed to [K][N]).
// Block tile 32(M) x 64(N), 8 waves (2x4), K-chunk 32, 1 WMMA/wave/chunk.
// Used for the tiny per-step GEMMs (M = 32).
// ---------------------------------------------------------------------------
template<bool OUT_BF16>
__global__ __launch_bounds__(256)
void gemm_wmma_k(const unsigned short* __restrict__ A0, int lda0,
                 const unsigned short* __restrict__ B0, int K0,
                 const unsigned short* __restrict__ A1, int lda1,
                 const unsigned short* __restrict__ B1, int K1,
                 const float* __restrict__ bias,
                 float* __restrict__ Cf, unsigned short* __restrict__ Cb,
                 int N)
{
    __shared__ unsigned short As[32 * LSTR];   // m-major
    __shared__ unsigned short Bs[64 * LSTR];   // n-major (transposed in LDS)

    const int tid  = threadIdx.x;
    const int lane = tid & 31;
    const int wave = tid >> 5;
    const int wm   = wave >> 2;      // 0..1
    const int wn   = wave & 3;       // 0..3
    const int m0   = blockIdx.y * 32;
    const int n0   = blockIdx.x * 64;

    v8f acc = {};

    const unsigned short* Ap = A0;
    const unsigned short* Bp = B0;
    int K = K0, lda = lda0;

    for (int pass = 0; pass < 2; ++pass) {
        if (pass == 1) {
            if (A1 == nullptr) break;
            Ap = A1; Bp = B1; K = K1; lda = lda1;
        }
        for (int kc = 0; kc < K; kc += 32) {
            {   // stage A chunk (32m x 32k): 4 bf16 per thread
                const int m  = tid & 31;
                const int kg = tid >> 5;                 // 0..7
                *(u32x2*)&As[m * LSTR + kg * 4] =
                    *(const u32x2*)&Ap[(size_t)(m0 + m) * lda + kc + kg * 4];
            }
            {   // stage B chunk (32k x 64n) transposed -> Bs[n][k]
                const int kr = tid & 31;
                const int ng = tid >> 5;                 // 0..7
                union { u32x4 q; unsigned short s[8]; } tb;
                tb.q = *(const u32x4*)&Bp[(size_t)(kc + kr) * N + n0 + ng * 8];
#pragma unroll
                for (int j = 0; j < 8; ++j)
                    Bs[(ng * 8 + j) * LSTR + kr] = tb.s[j];
            }
            if (kc + 32 < K)
                __builtin_prefetch(&Bp[(size_t)(kc + 32 + (tid & 31)) * N + n0], 0, 1);

            __syncthreads();
            {
                const int mrow = wm * 16 + (lane & 15);
                const int ncol = wn * 16 + (lane & 15);
                const int kh   = (lane >> 4) * 8;
                FragU a, b;
                a.q[0] = *(const u32x4*)&As[mrow * LSTR + kh];
                a.q[1] = *(const u32x4*)&As[mrow * LSTR + 16 + kh];
                b.q[0] = *(const u32x4*)&Bs[ncol * LSTR + kh];
                b.q[1] = *(const u32x4*)&Bs[ncol * LSTR + 16 + kh];
                acc = __builtin_amdgcn_wmma_f32_16x16x32_bf16(
                          false, a.v, false, b.v, (short)0, acc, false, false);
            }
            __syncthreads();
        }
    }

    const int nl = n0 + wn * 16 + (lane & 15);
    const int mb = m0 + wm * 16 + ((lane >> 4) << 3);
    const float bv = bias ? bias[nl] : 0.0f;
#pragma unroll
    for (int r = 0; r < 8; ++r) {
        const float v = acc[r] + bv;
        const int row = mb + r;
        if (OUT_BF16) Cb[(size_t)row * N + nl] = f2bf(v);
        else          Cf[(size_t)row * N + nl] = v;
    }
}

// ---------------------------------------------------------------------------
// Vocab-projection GEMM (the 67-GFLOP hot spot):
//   out[b][t][:] = H[t*32+b] @ out_W.T + out_b
// M=2048 (SEQ*BATCH), K=512, N=32000.
// Block tile 64(M) x 128(N); 8 waves 2(M)x4(N); each wave 32x32 via 2x2
// register tiling (4 WMMA / chunk, A/B frags reused). Double-buffered LDS,
// one barrier per K-chunk; A tile staged with async global->LDS b128.
// ---------------------------------------------------------------------------
__global__ __launch_bounds__(256)
void gemm_out_k(const unsigned short* __restrict__ H,    // [2048 x 512] bf16
                const unsigned short* __restrict__ Bt,   // [512 x 32000] bf16
                const float* __restrict__ bias,          // [32000]
                float* __restrict__ C)                   // [2048 x 32000] remapped
{
    __shared__ unsigned short As[2][64 * LSTR];          // 2 x 5.0 KB
    __shared__ unsigned short Bs[2][128 * LSTR];         // 2 x 10.0 KB

    const int tid  = threadIdx.x;
    const int lane = tid & 31;
    const int wave = tid >> 5;
    const int wm   = wave >> 2;                          // 0..1 -> +32 rows
    const int wn   = wave & 3;                           // 0..3 -> +32 cols
    const int m0   = blockIdx.y * 64;
    const int n0   = blockIdx.x * 128;

    // staging coordinates
    const int sa_m  = tid >> 2;                          // 0..63
    const int sa_kg = (tid & 3) * 8;                     // 0,8,16,24
    const int sb_kr = tid & 31;                          // 0..31
    const int sb_ng = tid >> 5;                          // 0..7 -> 16 cols each

    v8f acc00 = {}, acc01 = {}, acc10 = {}, acc11 = {};

    auto stage = [&](int buf, int kc) {
        // A tile 64m x 32k : one 16B transfer per lane
#ifdef USE_ASYNC_LDS
        async_copy16(&H[(size_t)(m0 + sa_m) * 512 + kc + sa_kg],
                     &As[buf][sa_m * LSTR + sa_kg]);
#else
        *(u32x4*)&As[buf][sa_m * LSTR + sa_kg] =
            *(const u32x4*)&H[(size_t)(m0 + sa_m) * 512 + kc + sa_kg];
#endif
        // B tile 32k x 128n : 16 bf16 per lane, transposed into Bs[n][k]
        union { u32x4 q; unsigned short s[8]; } t0, t1;
        const unsigned short* src = &Bt[(size_t)(kc + sb_kr) * VOCAB + n0 + sb_ng * 16];
        t0.q = *(const u32x4*)(src);
        t1.q = *(const u32x4*)(src + 8);
#pragma unroll
        for (int j = 0; j < 8; ++j) {
            Bs[buf][(sb_ng * 16 + j)     * LSTR + sb_kr] = t0.s[j];
            Bs[buf][(sb_ng * 16 + 8 + j) * LSTR + sb_kr] = t1.s[j];
        }
    };

    const int mrow = wm * 32 + (lane & 15);
    const int ncol = wn * 32 + (lane & 15);
    const int kh   = (lane >> 4) * 8;

    stage(0, 0);
#ifdef USE_ASYNC_LDS
    __builtin_amdgcn_s_wait_asynccnt(0);
#endif
    __syncthreads();

    int p = 0;
    for (int kc = 0; kc < 512; kc += 32) {
        if (kc + 32 < 512) stage(p ^ 1, kc + 32);        // overlaps with WMMA below

        const unsigned short* as = As[p];
        const unsigned short* bs = Bs[p];
        FragU a0, a1, b0, b1;
        a0.q[0] = *(const u32x4*)&as[mrow * LSTR + kh];
        a0.q[1] = *(const u32x4*)&as[mrow * LSTR + 16 + kh];
        a1.q[0] = *(const u32x4*)&as[(mrow + 16) * LSTR + kh];
        a1.q[1] = *(const u32x4*)&as[(mrow + 16) * LSTR + 16 + kh];
        b0.q[0] = *(const u32x4*)&bs[ncol * LSTR + kh];
        b0.q[1] = *(const u32x4*)&bs[ncol * LSTR + 16 + kh];
        b1.q[0] = *(const u32x4*)&bs[(ncol + 16) * LSTR + kh];
        b1.q[1] = *(const u32x4*)&bs[(ncol + 16) * LSTR + 16 + kh];

        acc00 = __builtin_amdgcn_wmma_f32_16x16x32_bf16(false, a0.v, false, b0.v, (short)0, acc00, false, false);
        acc01 = __builtin_amdgcn_wmma_f32_16x16x32_bf16(false, a0.v, false, b1.v, (short)0, acc01, false, false);
        acc10 = __builtin_amdgcn_wmma_f32_16x16x32_bf16(false, a1.v, false, b0.v, (short)0, acc10, false, false);
        acc11 = __builtin_amdgcn_wmma_f32_16x16x32_bf16(false, a1.v, false, b1.v, (short)0, acc11, false, false);

#ifdef USE_ASYNC_LDS
        __builtin_amdgcn_s_wait_asynccnt(0);
#endif
        __syncthreads();
        p ^= 1;
    }

    // epilogue: C/D layout (8 VGPRs; lanes 0-15 M=r, lanes 16-31 M=8+r),
    // H row m = t*32+b  ->  out row = b*64+t
    const int mb = m0 + wm * 32 + ((lane >> 4) << 3);
    const int nb = n0 + wn * 32 + (lane & 15);
#pragma unroll
    for (int i = 0; i < 2; ++i) {
#pragma unroll
        for (int j = 0; j < 2; ++j) {
            const v8f acc = (i == 0) ? (j == 0 ? acc00 : acc01)
                                     : (j == 0 ? acc10 : acc11);
            const int n  = nb + j * 16;
            const float bv = bias[n];
#pragma unroll
            for (int r = 0; r < 8; ++r) {
                const int m   = mb + i * 16 + r;
                const int row = (m & 31) * SEQ + (m >> 5);
                C[(size_t)row * VOCAB + n] = acc[r] + bv;
            }
        }
    }
}

// ---------------------------------------------------------------------------
// Setup kernels
// ---------------------------------------------------------------------------
__global__ void k_cvt_T(unsigned short* __restrict__ dst,
                        const float* __restrict__ src,
                        int rowsDst, int colsDst)
{
    int idx = blockIdx.x * 256 + threadIdx.x;
    if (idx >= rowsDst * colsDst) return;
    int k = idx / colsDst;
    int n = idx - k * colsDst;
    dst[idx] = f2bf(src[(size_t)n * rowsDst + k]);
}

__global__ void k_embed(unsigned short* __restrict__ emb,
                        const int* __restrict__ captions,
                        const float* __restrict__ embed_W)
{
    int idx = blockIdx.x * 256 + threadIdx.x;
    if (idx >= BATCH * SEQ * EMBED) return;
    int e  = idx & (EMBED - 1);
    int bt = idx >> 9;
    int tok = captions[bt];
    emb[idx] = f2bf(embed_W[(size_t)tok * EMBED + e] * 22.62741699796952f); // sqrt(512)
}

__global__ void k_fmean(float* __restrict__ fmean,
                        const float* __restrict__ features)
{
    int idx = blockIdx.x * 256 + threadIdx.x;
    if (idx >= BATCH * EMBED) return;
    int e = idx & (EMBED - 1);
    int b = idx >> 9;
    float s = 0.f;
    for (int r = 0; r < NREG; ++r)
        s += features[((size_t)b * NREG + r) * EMBED + e];
    fmean[idx] = s * (1.0f / NREG);
}

__global__ void k_init_hc(float* __restrict__ h_f32, float* __restrict__ c_f32,
                          unsigned short* __restrict__ h_bf16,
                          float* __restrict__ bias_sum,
                          const float* __restrict__ fmean,
                          const float* __restrict__ hW, const float* __restrict__ hb,
                          const float* __restrict__ cW, const float* __restrict__ cb,
                          const float* __restrict__ bih, const float* __restrict__ bhh)
{
    int idx = blockIdx.x * 256 + threadIdx.x;
    if (idx >= BATCH * HIDDEN) return;
    int j = idx & (HIDDEN - 1);
    int b = idx >> 9;
    const float* fm = fmean + (size_t)b * EMBED;
    float h = hb[j], c = cb[j];
    for (int e = 0; e < EMBED; ++e) {
        h = fmaf(fm[e], hW[(size_t)j * EMBED + e], h);
        c = fmaf(fm[e], cW[(size_t)j * EMBED + e], c);
    }
    h_f32[idx] = h;
    c_f32[idx] = c;
    h_bf16[idx] = f2bf(h);
    if (idx < GATES4) bias_sum[idx] = bih[idx] + bhh[idx];
}

// ---------------------------------------------------------------------------
// Per-step: LSTM cell (for step t-1) + attention + build xcat for step t.
// One workgroup per batch row. t == 64 -> final cell update only.
// ---------------------------------------------------------------------------
__global__ __launch_bounds__(256)
void k_step_pre(const float* __restrict__ features,
                const unsigned short* __restrict__ emb_bf16,
                const float* __restrict__ gates,
                float* __restrict__ h_f32, float* __restrict__ c_f32,
                unsigned short* __restrict__ h_bf16,
                unsigned short* __restrict__ H_bf16,
                unsigned short* __restrict__ xcat_bf16,
                int t)
{
    const int b   = blockIdx.x;
    const int tid = threadIdx.x;
    __shared__ float sh_h[HIDDEN];
    __shared__ float sh_s[NREG];
    __shared__ float sh_red[64];

    if (t > 0) {
        for (int j = tid; j < HIDDEN; j += 256) {
            const size_t g0 = (size_t)b * GATES4 + j;
            float ig = gates[g0];
            float fg = gates[g0 + 512];
            float gg = gates[g0 + 1024];
            float og = gates[g0 + 1536];
            float si = 1.f / (1.f + expf(-ig));
            float sf = 1.f / (1.f + expf(-fg));
            float so = 1.f / (1.f + expf(-og));
            float c  = sf * c_f32[b * HIDDEN + j] + si * tanhf(gg);
            float h  = so * tanhf(c);
            c_f32[b * HIDDEN + j] = c;
            h_f32[b * HIDDEN + j] = h;
            sh_h[j] = h;
            unsigned short hb = f2bf(h);
            h_bf16[b * HIDDEN + j] = hb;
            H_bf16[((size_t)(t - 1) * BATCH + b) * HIDDEN + j] = hb;
        }
    } else {
        for (int j = tid; j < HIDDEN; j += 256) sh_h[j] = h_f32[b * HIDDEN + j];
    }
    __syncthreads();
    if (t >= SEQ) return;

    if (tid < NREG) {
        const float* fr = features + ((size_t)b * NREG + tid) * EMBED;
        float s = 0.f;
        for (int e = 0; e < EMBED; ++e) s = fmaf(fr[e], sh_h[e], s);
        sh_s[tid] = s;
    }
    __syncthreads();

    if (tid < 32) {
        float m = -3.0e38f;
        for (int r = tid; r < NREG; r += 32) m = fmaxf(m, sh_s[r]);
        sh_red[tid] = m;
    }
    __syncthreads();
    if (tid == 0) {
        float m = sh_red[0];
        for (int i = 1; i < 32; ++i) m = fmaxf(m, sh_red[i]);
        sh_red[32] = m;
    }
    __syncthreads();
    const float mx = sh_red[32];
    if (tid < NREG) sh_s[tid] = expf(sh_s[tid] - mx);
    __syncthreads();
    if (tid < 32) {
        float s = 0.f;
        for (int r = tid; r < NREG; r += 32) s += sh_s[r];
        sh_red[tid] = s;
    }
    __syncthreads();
    if (tid == 0) {
        float s = 0.f;
        for (int i = 0; i < 32; ++i) s += sh_red[i];
        sh_red[33] = 1.f / s;
    }
    __syncthreads();
    const float inv = sh_red[33];

    for (int e = tid; e < EMBED; e += 256) {
        float cx = 0.f;
        for (int r = 0; r < NREG; ++r)
            cx = fmaf(sh_s[r], features[((size_t)b * NREG + r) * EMBED + e], cx);
        cx *= inv;
        xcat_bf16[(size_t)b * 1024 + e]       = f2bf(cx);
        xcat_bf16[(size_t)b * 1024 + 512 + e] =
            emb_bf16[((size_t)b * SEQ + t) * EMBED + e];
    }
}

// ---------------------------------------------------------------------------
// Host launcher
// ---------------------------------------------------------------------------
extern "C" void kernel_launch(void* const* d_in, const int* in_sizes, int n_in,
                              void* d_out, int out_size, void* d_ws, size_t ws_size,
                              hipStream_t stream)
{
    const float* features  = (const float*)d_in[0];
    const int*   captions  = (const int*)  d_in[1];
    const float* embed_W   = (const float*)d_in[2];
    const float* init_h_W  = (const float*)d_in[3];
    const float* init_h_b  = (const float*)d_in[4];
    const float* init_c_W  = (const float*)d_in[5];
    const float* init_c_b  = (const float*)d_in[6];
    const float* reshape_W = (const float*)d_in[7];
    const float* reshape_b = (const float*)d_in[8];
    const float* lstm_Wih  = (const float*)d_in[9];
    const float* lstm_Whh  = (const float*)d_in[10];
    const float* lstm_bih  = (const float*)d_in[11];
    const float* lstm_bhh  = (const float*)d_in[12];
    const float* out_W     = (const float*)d_in[13];
    const float* out_b     = (const float*)d_in[14];
    float* out = (float*)d_out;

    char* ws = (char*)d_ws;
    size_t off = 0;
    auto alloc = [&](size_t bytes) -> void* {
        void* p = ws + off;
        off = (off + bytes + 255) & ~(size_t)255;
        return p;
    };
    unsigned short* outWt     = (unsigned short*)alloc((size_t)EMBED * VOCAB * 2);
    unsigned short* WihT      = (unsigned short*)alloc((size_t)EMBED * GATES4 * 2);
    unsigned short* WhhT      = (unsigned short*)alloc((size_t)HIDDEN * GATES4 * 2);
    unsigned short* reshapeWt = (unsigned short*)alloc((size_t)1024 * EMBED * 2);
    unsigned short* emb_bf16  = (unsigned short*)alloc((size_t)BATCH * SEQ * EMBED * 2);
    unsigned short* H_bf16    = (unsigned short*)alloc((size_t)SEQ * BATCH * HIDDEN * 2);
    float*          h_f32     = (float*)alloc((size_t)BATCH * HIDDEN * 4);
    float*          c_f32     = (float*)alloc((size_t)BATCH * HIDDEN * 4);
    unsigned short* h_bf16    = (unsigned short*)alloc((size_t)BATCH * HIDDEN * 2);
    float*          fmean     = (float*)alloc((size_t)BATCH * EMBED * 4);
    unsigned short* xcat_bf16 = (unsigned short*)alloc((size_t)BATCH * 1024 * 2);
    unsigned short* x_bf16    = (unsigned short*)alloc((size_t)BATCH * EMBED * 2);
    float*          gates     = (float*)alloc((size_t)BATCH * GATES4 * 4);
    float*          bias_sum  = (float*)alloc((size_t)GATES4 * 4);

    auto g1 = [](size_t n) { return dim3((unsigned)((n + 255) / 256)); };

    k_cvt_T<<<g1((size_t)EMBED * VOCAB),   256, 0, stream>>>(outWt,     out_W,     EMBED, VOCAB);
    k_cvt_T<<<g1((size_t)EMBED * GATES4),  256, 0, stream>>>(WihT,      lstm_Wih,  EMBED, GATES4);
    k_cvt_T<<<g1((size_t)HIDDEN * GATES4), 256, 0, stream>>>(WhhT,      lstm_Whh,  HIDDEN, GATES4);
    k_cvt_T<<<g1((size_t)1024 * EMBED),    256, 0, stream>>>(reshapeWt, reshape_W, 1024, EMBED);
    k_embed<<<g1((size_t)BATCH * SEQ * EMBED), 256, 0, stream>>>(emb_bf16, captions, embed_W);
    k_fmean<<<g1((size_t)BATCH * EMBED), 256, 0, stream>>>(fmean, features);
    k_init_hc<<<g1((size_t)BATCH * HIDDEN), 256, 0, stream>>>(
        h_f32, c_f32, h_bf16, bias_sum, fmean,
        init_h_W, init_h_b, init_c_W, init_c_b, lstm_bih, lstm_bhh);

    for (int t = 0; t < SEQ; ++t) {
        k_step_pre<<<dim3(BATCH), 256, 0, stream>>>(
            features, emb_bf16, gates, h_f32, c_f32, h_bf16, H_bf16, xcat_bf16, t);
        // x = xcat @ reshape_W.T + reshape_b   [32 x 512], K=1024 -> bf16
        gemm_wmma_k<true><<<dim3(EMBED / 64, 1), 256, 0, stream>>>(
            xcat_bf16, 1024, reshapeWt, 1024,
            nullptr, 0, nullptr, 0,
            reshape_b, nullptr, x_bf16, EMBED);
        // gates = x @ Wih.T + h @ Whh.T + (bih+bhh)   [32 x 2048] -> f32
        gemm_wmma_k<false><<<dim3(GATES4 / 64, 1), 256, 0, stream>>>(
            x_bf16, EMBED, WihT, EMBED,
            h_bf16, HIDDEN, WhhT, HIDDEN,
            bias_sum, gates, nullptr, GATES4);
    }
    k_step_pre<<<dim3(BATCH), 256, 0, stream>>>(
        features, emb_bf16, gates, h_f32, c_f32, h_bf16, H_bf16, xcat_bf16, SEQ);

    // vocab projection: 64x128 tiles, double-buffered, async-LDS staged
    gemm_out_k<<<dim3(VOCAB / 128, (SEQ * BATCH) / 64), 256, 0, stream>>>(
        H_bf16, outWt, out_b, out);
}
